// RoIHeads_24807731102261
// MI455X (gfx1250) — compile-verified, matrix operand order checked by reading, not run on hardware
//
#include <hip/hip_runtime.h>
#include <hip/hip_bf16.h>

typedef __attribute__((ext_vector_type(16))) _Float16 v16h;
typedef __attribute__((ext_vector_type(8)))  float    v8f;
typedef __attribute__((ext_vector_type(4)))  unsigned int u32x4;
typedef __attribute__((ext_vector_type(8)))  int      i32x8;
typedef __attribute__((ext_vector_type(4)))  int      i32x4;

#define NPROP   65536
#define NCLS    81
#define NFG     80          // foreground classes 1..80
#define KKEEP   100
#define MAXCAND 3072        // expected ~1900 candidates/class at 2.9% pass rate
#define SCORE_THRESH 0.05f
#define NMS_THRESH   0.5f
#define MIN_SIZE     1.0f
#define BBOX_CLIP    4.135166556742356f   // log(1000/16)
#define ROWS_PER_BLOCK 128                // 8 wave32 x 16 rows
#define OUT_BOXES   (NFG * KKEEP * 4)     // 32000
#define OUT_SCORES  (NFG * KKEEP)         // 8000
#define OUT_TOTAL   (OUT_BOXES + 2 * OUT_SCORES) // 48000

// ---------------------------------------------------------------------------
// Kernel 0: zero outputs + per-class candidate counters (deterministic state)
// ---------------------------------------------------------------------------
__global__ void rh_zero_kernel(float* out, int* counts) {
    int i = blockIdx.x * blockDim.x + threadIdx.x;
    if (i < OUT_TOTAL) out[i] = 0.0f;
    if (i < NFG)       counts[i] = 0;
}

// ---------------------------------------------------------------------------
// Kernel 1: softmax (WMMA row-sum GEMV) + box decode/clip + candidate compact
// Block = 256 threads = 8 waves; each wave owns 16 proposal rows.
// ---------------------------------------------------------------------------
__global__ void __launch_bounds__(256)
rh_decode_kernel(const float* __restrict__ logits,
                 const float* __restrict__ breg,
                 const float* __restrict__ prop,
                 const int* __restrict__ imh_p,
                 const int* __restrict__ imw_p,
                 int* __restrict__ counts,
                 float* __restrict__ cand) {
    __shared__ float s_sum[ROWS_PER_BLOCK];
    __shared__ float s_max[ROWS_PER_BLOCK];

    const int tid   = threadIdx.x;
    const int lane  = tid & 31;
    const int wave  = tid >> 5;
    const int rbase = blockIdx.x * ROWS_PER_BLOCK + wave * 16;

    // ---- stage 1: per-row max, then row-sum of exp() via WMMA ----
    // Wave handles 16 rows: lanes L and L+16 both serve row rbase + (L%16),
    // covering the two K-halves of the 16-bit A-matrix layout (ISA 7.12.2).
    {
        const int myrow = rbase + (lane & 15);
        const float* lr = logits + (size_t)myrow * NCLS;
        float m = -3.0e38f;
        for (int k = 0; k < NCLS; ++k) m = fmaxf(m, lr[k]);

        v16h ones;
        for (int e = 0; e < 16; ++e) ones[e] = (_Float16)1.0f;
        v8f c = {};
        const int off = (lane < 16) ? 0 : 8;
        for (int chunk = 0; chunk < 3; ++chunk) {   // K = 96 covers 81 classes
            v16h a;
            for (int e = 0; e < 16; ++e) {
                int k = chunk * 32 + off + ((e < 8) ? e : (e + 8));
                float v = (k < NCLS) ? __expf(lr[k] - m) : 0.0f;
                a[e] = (_Float16)v;
            }
            // D[m][n] += sum_k A[m][k] * 1  ->  every column holds the row sum
            c = __builtin_amdgcn_wmma_f32_16x16x32_f16(
                    false, a, false, ones, (short)0, c, false, false);
        }
        // C/D layout: lane 0 VGPR v = D[v][0] (rows 0..7); lane 16 = rows 8..15
        if (lane == 0)  for (int v = 0; v < 8; ++v) s_sum[wave * 16 + v]     = c[v];
        if (lane == 16) for (int v = 0; v < 8; ++v) s_sum[wave * 16 + 8 + v] = c[v];
        if (lane < 16)  s_max[wave * 16 + lane] = m;
    }
    __syncthreads();

    // ---- stage 2: scores + decode + clip + compaction (2 threads / row) ----
    const int rlocal = tid & (ROWS_PER_BLOCK - 1);
    const int half   = tid >> 7;                 // 0 -> classes 1..40, 1 -> 41..80
    const int row    = blockIdx.x * ROWS_PER_BLOCK + rlocal;
    const float mx   = s_max[rlocal];
    const float inv_sum = 1.0f / s_sum[rlocal];
    const float W = (float)imw_p[0];
    const float H = (float)imh_p[0];

    const float px1 = prop[(size_t)row * 4 + 0];
    const float py1 = prop[(size_t)row * 4 + 1];
    const float px2 = prop[(size_t)row * 4 + 2];
    const float py2 = prop[(size_t)row * 4 + 3];
    const float pw = px2 - px1, ph = py2 - py1;
    const float pcx = px1 + 0.5f * pw, pcy = py1 + 0.5f * ph;

    const float* lr = logits + (size_t)row * NCLS;
    const float* dr = breg + (size_t)row * (NCLS * 4);

    const int c0 = 1 + half * 40;
    for (int c = c0; c < c0 + 40; ++c) {
        float score = __expf(lr[c] - mx) * inv_sum;
        if (score < SCORE_THRESH) continue;
        float dx = dr[c * 4 + 0] * 0.1f;
        float dy = dr[c * 4 + 1] * 0.1f;
        float dw = fminf(dr[c * 4 + 2] * 0.2f, BBOX_CLIP);
        float dh = fminf(dr[c * 4 + 3] * 0.2f, BBOX_CLIP);
        float cx = dx * pw + pcx, cy = dy * ph + pcy;
        float bw = __expf(dw) * pw, bh = __expf(dh) * ph;
        float x1 = fminf(fmaxf(cx - 0.5f * bw, 0.0f), W);
        float y1 = fminf(fmaxf(cy - 0.5f * bh, 0.0f), H);
        float x2 = fminf(fmaxf(cx + 0.5f * bw, 0.0f), W);
        float y2 = fminf(fmaxf(cy + 0.5f * bh, 0.0f), H);
        if ((x2 - x1) < MIN_SIZE || (y2 - y1) < MIN_SIZE) continue;
        int idx = atomicAdd(&counts[c - 1], 1);
        if (idx < MAXCAND) {
            float* rec = cand + ((size_t)(c - 1) * MAXCAND + idx) * 5;
            rec[0] = x1; rec[1] = y1; rec[2] = x2; rec[3] = y2; rec[4] = score;
        }
    }
}

// ---------------------------------------------------------------------------
// Kernel 2: per-class greedy NMS in LDS. One block per class.
// Candidate list staged into LDS with the Tensor Data Mover (TDM).
// ---------------------------------------------------------------------------
__global__ void __launch_bounds__(256)
rh_nms_kernel(const float* __restrict__ cand,
              const int* __restrict__ counts,
              float* __restrict__ out) {
    __shared__ float lcand[MAXCAND * 5];   // 60 KB: {x1,y1,x2,y2,score} AoS
    __shared__ float rv[256];
    __shared__ int   ri[256];

    const int cls  = blockIdx.x;           // 0..79 -> label cls+1
    const int tid  = threadIdx.x;
    const int cnt  = min(counts[cls], MAXCAND);

    // --- TDM: DMA cnt*5 floats of this class's candidates into LDS ---
    if (tid < 32) {                        // issue from wave 0 only
        unsigned n = (unsigned)cnt * 5u;   // elements (<= 15360, fits 16-bit tile dim)
        unsigned lds_off = (unsigned)(unsigned long long)(void*)lcand;
        unsigned long long ga =
            (unsigned long long)(const void*)(cand + (size_t)cls * MAXCAND * 5);
        // D# group 0: count=1 | lds_addr | global_addr[56:0] | type=2
        u32x4 g0 = { 1u, lds_off,
                     (unsigned)(ga & 0xFFFFFFFFull),
                     (unsigned)((ga >> 32) & 0x1FFFFFFull) | 0x80000000u };
        // D# group 1: data_size=4B; tensor_dim0=n, tensor_dim1=1;
        //             tile_dim0=n, tile_dim1=1; tensor_dim0_stride=n
        i32x8 g1 = { (int)0x00020000u,
                     (int)((n & 0xFFFFu) << 16),
                     (int)(((n >> 16) & 0xFFFFu) | (1u << 16)),
                     (int)((n & 0xFFFFu) << 16),
                     1,
                     (int)n,
                     0, 0 };
        i32x4 g2 = { 0, 0, 0, 0 };
        i32x4 g3 = { 0, 0, 0, 0 };
#if defined(__clang_major__) && (__clang_major__ >= 23)
        i32x8 g4 = { 0, 0, 0, 0, 0, 0, 0, 0 };
        __builtin_amdgcn_tensor_load_to_lds(g0, g1, g2, g3, g4, 0);
#else
        __builtin_amdgcn_tensor_load_to_lds(g0, g1, g2, g3, 0);
#endif
        __builtin_amdgcn_s_wait_tensorcnt(0);
    }
    __syncthreads();

    float* outb = out;                         // [8000][4]
    float* outs = out + OUT_BOXES;             // [8000]
    int*   outl = (int*)out + OUT_BOXES + OUT_SCORES;

    for (int it = 0; it < KKEEP; ++it) {
        // block-wide argmax over live scores
        float bv = -1.0e30f; int bi = -1;
        for (int j = tid; j < cnt; j += 256) {
            float v = lcand[j * 5 + 4];
            if (v > bv) { bv = v; bi = j; }
        }
        rv[tid] = bv; ri[tid] = bi;
        __syncthreads();
        for (int off = 128; off > 0; off >>= 1) {
            if (tid < off && rv[tid + off] > rv[tid]) {
                rv[tid] = rv[tid + off]; ri[tid] = ri[tid + off];
            }
            __syncthreads();
        }
        float best = rv[0]; int bidx = ri[0];
        if (best < 0.0f) break;                // uniform: list exhausted

        float bx1 = lcand[bidx * 5 + 0], by1 = lcand[bidx * 5 + 1];
        float bx2 = lcand[bidx * 5 + 2], by2 = lcand[bidx * 5 + 3];
        float barea = (bx2 - bx1) * (by2 - by1);
        if (tid == 0) {
            int o = cls * KKEEP + it;
            outb[o * 4 + 0] = bx1; outb[o * 4 + 1] = by1;
            outb[o * 4 + 2] = bx2; outb[o * 4 + 3] = by2;
            outs[o] = best;
            outl[o] = cls + 1;
            lcand[bidx * 5 + 4] = -1.0f;
        }
        __syncthreads();
        // suppress IoU > 0.5
        for (int j = tid; j < cnt; j += 256) {
            float s = lcand[j * 5 + 4];
            if (s < 0.0f) continue;
            float x1 = lcand[j * 5 + 0], y1 = lcand[j * 5 + 1];
            float x2 = lcand[j * 5 + 2], y2 = lcand[j * 5 + 3];
            float iw = fminf(bx2, x2) - fmaxf(bx1, x1);
            float ih = fminf(by2, y2) - fmaxf(by1, y1);
            float inter = fmaxf(iw, 0.0f) * fmaxf(ih, 0.0f);
            float area  = (x2 - x1) * (y2 - y1);
            float iou = inter / fmaxf(barea + area - inter, 1e-9f);
            if (iou > NMS_THRESH) lcand[j * 5 + 4] = -1.0f;
        }
        __syncthreads();
    }
}

// ---------------------------------------------------------------------------
extern "C" void kernel_launch(void* const* d_in, const int* in_sizes, int n_in,
                              void* d_out, int out_size, void* d_ws, size_t ws_size,
                              hipStream_t stream) {
    const float* logits = (const float*)d_in[0];   // [65536, 81]
    const float* breg   = (const float*)d_in[1];   // [65536, 324]
    const float* prop   = (const float*)d_in[2];   // [65536, 4]
    const int*   imh    = (const int*)d_in[3];
    const int*   imw    = (const int*)d_in[4];
    float* out = (float*)d_out;

    // workspace layout: [0,1024) counts (80 ints, padded) ; then candidates
    int*   counts = (int*)d_ws;
    float* cand   = (float*)((char*)d_ws + 1024);  // 80*3072*5 floats = 4.7 MB

    rh_zero_kernel<<<(OUT_TOTAL + 255) / 256, 256, 0, stream>>>(out, counts);
    rh_decode_kernel<<<NPROP / ROWS_PER_BLOCK, 256, 0, stream>>>(
        logits, breg, prop, imh, imw, counts, cand);
    rh_nms_kernel<<<NFG, 256, 0, stream>>>(cand, counts, out);
}
